// MultiHeadMLPAttention_45767171506729
// MI455X (gfx1250) — matrix operand order repaired and checked by reading, not compiled, not run on hardware
//
#include <hip/hip_runtime.h>
#include <hip/hip_bf16.h>
#include <math.h>

// ---------------------------------------------------------------------------
// MultiHeadMLPAttention fused for MI455X (gfx1250, wave32, WMMA).
// Dominant GEMM (key@Wk, 65536x1024x1024) runs on v_wmma_f32_16x16x32_bf16
// with the tanh(q+k).w_score reduction fused into the epilogue so key_up is
// never materialized. HBM traffic ~= key(256MB)+value(256MB) => ~22us floor.
// ---------------------------------------------------------------------------

typedef __bf16 bf16_t;
typedef __attribute__((ext_vector_type(16))) bf16_t v16bf;
typedef __attribute__((ext_vector_type(8)))  float  v8f;

#define B_SZ   32
#define LK     2048
#define DM     1024
#define H_HEADS 16

__device__ __forceinline__ unsigned short f2bf(float f) {
  unsigned int u = __float_as_uint(f);
  u += 0x7FFFu + ((u >> 16) & 1u);   // round-to-nearest-even
  return (unsigned short)(u >> 16);
}

union Frag {
  v16bf v;
  unsigned short u[16];
  uint4 q[2];
};

// ---------------------------------------------------------------------------
// Kernel 0: pack Wk (f32, [K=1024, N=1024] row-major) into bf16 WMMA
// B-fragment layout: [ntile(64)][ktile(32)][lane(32)][e(16)] bf16.
// B layout (32x16 bf16, contiguous k order chosen here, same as A): lanes 0-15 hold K=0..15
// for column n=lane, lanes 16-31 hold K=16..31; 2 consecutive K per VGPR.
// => element e of lane: k = kt*32 + (lane>>4)*16 + e, n = nt*16 + (lane&15).
// ---------------------------------------------------------------------------
__global__ void pack_wk_kernel(const float* __restrict__ Wk,
                               uint4* __restrict__ out) {
  int idx  = blockIdx.x * blockDim.x + threadIdx.x;   // 64*32*32 = 65536
  int lane = idx & 31;
  int kt   = (idx >> 5) & 31;
  int nt   = idx >> 10;
  int half = lane >> 4;
  int n    = nt * 16 + (lane & 15);
  Frag f;
#pragma unroll
  for (int e = 0; e < 16; ++e) {
    int k = kt * 32 + half * 16 + e;
    f.u[e] = f2bf(Wk[(size_t)k * DM + n]);
  }
  out[(size_t)idx * 2 + 0] = f.q[0];
  out[(size_t)idx * 2 + 1] = f.q[1];
}

// ---------------------------------------------------------------------------
// Kernel 1: q_up[b,n] = query[b,:] @ Wq[:,n] + bq[n]   (tiny: 32x1024x1024)
// ---------------------------------------------------------------------------
__global__ void qup_kernel(const float* __restrict__ query,
                           const float* __restrict__ Wq,
                           const float* __restrict__ bq,
                           float* __restrict__ qup) {
  int b = blockIdx.x;
  int n = blockIdx.y * blockDim.x + threadIdx.x;
  const float* q = query + (size_t)b * DM;
  float acc = bq[n];
  for (int k = 0; k < DM; ++k)
    acc += q[k] * Wq[(size_t)k * DM + n];
  qup[(size_t)b * DM + n] = acc;
}

// ---------------------------------------------------------------------------
// Kernel 2: fused scores. One WG = 32 key rows of one batch (2 row-tiles of
// 16). 4 waves; wave w owns heads 4w..4w+3 exclusively (no atomics).
// Phase 1: key rows f32 -> bf16 A-fragments in LDS (64KB).
// A layout (16x32 bf16): lane = (half<<4)|m; element e holds
// K = half*16 + e.
// Phase 2: per N-tile accumulate C over K=1024 (32 wmma steps), two row-tiles
// share each B fragment; epilogue: tanh(c + bk[n] + qup[b,n]) * w_score[n%64],
// reduced across the 16-lane halves (C layout: VGPR r -> M=r / M=r+8).
// ---------------------------------------------------------------------------
__global__ __launch_bounds__(128)
void scores_kernel(const float* __restrict__ key,
                   const uint4* __restrict__ wkp,
                   const float* __restrict__ bk,
                   const float* __restrict__ qup,
                   const float* __restrict__ wsc,
                   float* __restrict__ scores) {
  __shared__ uint4 ldsA[4096];   // 2 rt * 32 kt * 32 lane * 32B = 64 KB
  const int b  = blockIdx.y;
  const int l0 = blockIdx.x * 32;
  const int t  = threadIdx.x;

  // ---- Phase 1: stage + convert 32 rows x 1024 K (coalesced float2 reads)
  const float* krows = key + ((size_t)b * LK + l0) * DM;
  unsigned int* ldsw = (unsigned int*)ldsA;
  for (int p = t; p < 32 * 512; p += 128) {
    int rl = p >> 9;              // local row 0..31
    int k  = (p & 511) * 2;       // even k, pair (k,k+1) shares a lane
    float2 xy = *(const float2*)(krows + (size_t)rl * DM + k);
    int kt = k >> 5, km = k & 31;
    int half = km >> 4;
    int e    = km & 15;
    int lane = (half << 4) + (rl & 15);
    int rt   = rl >> 4;
    unsigned int pk = (unsigned int)f2bf(xy.x) | ((unsigned int)f2bf(xy.y) << 16);
    ldsw[(((rt * 32 + kt) * 32 + lane) * 16 + e) >> 1] = pk;
  }
  __syncthreads();

  // ---- Phase 2: WMMA + fused tanh-dot epilogue
  const int w = t >> 5, lane = t & 31;
  const float* qb = qup + (size_t)b * DM;
  for (int hi = 0; hi < 4; ++hi) {
    int h = w * 4 + hi;
    float s0[8], s1[8];
#pragma unroll
    for (int r = 0; r < 8; ++r) { s0[r] = 0.f; s1[r] = 0.f; }
    for (int j = 0; j < 4; ++j) {            // 4 N-tiles per head (64 cols)
      int nt = h * 4 + j;
      int n  = nt * 16 + (lane & 15);
      float bqn = bk[n] + qb[n];
      float wn  = wsc[n & 63];
      v8f c0 = {}, c1 = {};
      const uint4* bp = wkp + (size_t)nt * 2048;   // 32kt*32lane*2 uint4
#pragma unroll 4
      for (int kt = 0; kt < 32; ++kt) {
        Frag bf_, a0, a1;
        int bi = (kt * 32 + lane) * 2;
        bf_.q[0] = bp[bi];            bf_.q[1] = bp[bi + 1];
        a0.q[0]  = ldsA[bi];          a0.q[1]  = ldsA[bi + 1];
        a1.q[0]  = ldsA[2048 + bi];   a1.q[1]  = ldsA[2048 + bi + 1];
        c0 = __builtin_amdgcn_wmma_f32_16x16x32_bf16(false, a0.v, false, bf_.v,
                                                     (short)0, c0, false, false);
        c1 = __builtin_amdgcn_wmma_f32_16x16x32_bf16(false, a1.v, false, bf_.v,
                                                     (short)0, c1, false, false);
      }
#pragma unroll
      for (int r = 0; r < 8; ++r) {
        s0[r] += tanhf(c0[r] + bqn) * wn;
        s1[r] += tanhf(c1[r] + bqn) * wn;
      }
    }
    // reduce across the 16-lane halves (xor 1,2,4,8 stays within a half)
    float* srow = scores + ((size_t)b * H_HEADS + h) * LK + l0;
#pragma unroll
    for (int r = 0; r < 8; ++r) {
      float v0 = s0[r], v1 = s1[r];
#pragma unroll
      for (int off = 8; off >= 1; off >>= 1) {
        v0 += __shfl_xor(v0, off);
        v1 += __shfl_xor(v1, off);
      }
      if ((lane & 15) == 0) {
        int m = r + (lane >> 4) * 8;    // lane0 -> M=r, lane16 -> M=r+8
        srow[m]      = v0;              // row-tile 0
        srow[16 + m] = v1;              // row-tile 1
      }
    }
  }
}

// ---------------------------------------------------------------------------
// Kernel 3: masked softmax over Lk per (b,h); adds b_score; writes attn.
// ---------------------------------------------------------------------------
__global__ void softmax_kernel(const float* __restrict__ scores,
                               const unsigned char* __restrict__ mask, // jax bool = 1B
                               const float* __restrict__ b_score,
                               float* __restrict__ attn) {
  __shared__ float red[256];
  int bh = blockIdx.x;
  int b  = bh >> 4;
  int t  = threadIdx.x;
  const float* src = scores + (size_t)bh * LK;
  const unsigned char* mk = mask + (size_t)b * LK;
  float bs = b_score[0];
  float vals[8];
  float mx = -INFINITY;
#pragma unroll
  for (int i = 0; i < 8; ++i) {
    int l = t + i * 256;
    float v = mk[l] ? -INFINITY : (src[l] + bs);
    vals[i] = v;
    mx = fmaxf(mx, v);
  }
  red[t] = mx; __syncthreads();
  for (int s = 128; s > 0; s >>= 1) {
    if (t < s) red[t] = fmaxf(red[t], red[t + s]);
    __syncthreads();
  }
  mx = red[0]; __syncthreads();
  float sum = 0.f;
#pragma unroll
  for (int i = 0; i < 8; ++i) {
    float e = __expf(vals[i] - mx);
    vals[i] = e; sum += e;
  }
  red[t] = sum; __syncthreads();
  for (int s = 128; s > 0; s >>= 1) {
    if (t < s) red[t] += red[t + s];
    __syncthreads();
  }
  float inv = 1.f / red[0];
  float* dst = attn + (size_t)bh * LK;
#pragma unroll
  for (int i = 0; i < 8; ++i) dst[t + i * 256] = vals[i] * inv;
}

__global__ void zero_kernel(float* __restrict__ p, int n) {
  int i = blockIdx.x * blockDim.x + threadIdx.x;
  if (i < n) p[i] = 0.f;
}

// ---------------------------------------------------------------------------
// Kernel 4: context[b, h*64+dv] = sum_l attn[b,h,l] * value[b,l,h*64+dv].
// Streams value (256 MB) fully coalesced; 64-l chunks, f32 atomics into ws.
// ---------------------------------------------------------------------------
__global__ void context_kernel(const float* __restrict__ attn,
                               const float* __restrict__ value,
                               float* __restrict__ ctx) {
  int b = blockIdx.x, ch = blockIdx.y;     // 32 chunks of 64 keys
  int col = threadIdx.x;                   // 0..1023
  int h = col >> 6;
  const float* ap = attn  + ((size_t)b * H_HEADS + h) * LK + ch * 64;
  const float* vp = value + ((size_t)b * LK + (size_t)ch * 64) * DM + col;
  float acc = 0.f;
#pragma unroll 4
  for (int li = 0; li < 64; ++li)
    acc += ap[li] * vp[(size_t)li * DM];
  atomicAdd(&ctx[(size_t)b * DM + col], acc);
}

// ---------------------------------------------------------------------------
// Kernel 5: output = context @ Wf + bf   (32x1024x1024, trivial)
// ---------------------------------------------------------------------------
__global__ void output_kernel(const float* __restrict__ ctx,
                              const float* __restrict__ Wf,
                              const float* __restrict__ bf,
                              float* __restrict__ out) {
  int b = blockIdx.x;
  int n = blockIdx.y * blockDim.x + threadIdx.x;
  const float* c = ctx + (size_t)b * DM;
  float acc = bf[n];
  for (int k = 0; k < DM; ++k)
    acc += c[k] * Wf[(size_t)k * DM + n];
  out[(size_t)b * DM + n] = acc;
}

// ---------------------------------------------------------------------------
extern "C" void kernel_launch(void* const* d_in, const int* in_sizes, int n_in,
                              void* d_out, int out_size, void* d_ws, size_t ws_size,
                              hipStream_t stream) {
  (void)in_sizes; (void)n_in; (void)out_size; (void)ws_size;
  const float* key   = (const float*)d_in[0];
  const float* value = (const float*)d_in[1];
  const float* query = (const float*)d_in[2];
  const unsigned char* mask = (const unsigned char*)d_in[3];  // bool -> 1 byte
  const float* Wk  = (const float*)d_in[4];
  const float* bk  = (const float*)d_in[5];
  const float* Wq  = (const float*)d_in[6];
  const float* bq  = (const float*)d_in[7];
  const float* wsc = (const float*)d_in[8];
  const float* b_score = (const float*)d_in[9];
  const float* Wf  = (const float*)d_in[10];
  const float* bf  = (const float*)d_in[11];

  // workspace layout (total ~6.3 MB)
  char* ws = (char*)d_ws;
  uint4* wkp   = (uint4*)ws;                                          // 2 MB
  float* qup   = (float*)(ws + (2u << 20));                           // 128 KB
  float* scores= (float*)(ws + (2u << 20) + (128u << 10));            // 4 MB
  float* ctx   = (float*)(ws + (6u << 20) + (128u << 10));            // 128 KB

  float* out  = (float*)d_out;                 // [32,1,1024]
  float* attn = out + (size_t)B_SZ * DM;       // [32,16,1,2048]

  pack_wk_kernel <<<256, 256, 0, stream>>>(Wk, wkp);
  qup_kernel     <<<dim3(32, 4), 256, 0, stream>>>(query, Wq, bq, qup);
  scores_kernel  <<<dim3(LK / 32, B_SZ), 128, 0, stream>>>(key, wkp, bk, qup, wsc, scores);
  softmax_kernel <<<B_SZ * H_HEADS, 256, 0, stream>>>(scores, mask, b_score, attn);
  zero_kernel    <<<(B_SZ * DM + 255) / 256, 256, 0, stream>>>(ctx, B_SZ * DM);
  context_kernel <<<dim3(B_SZ, 32), 1024, 0, stream>>>(attn, value, ctx);
  output_kernel  <<<dim3(B_SZ, 4), 256, 0, stream>>>(ctx, Wf, bf, out);
}
